// ContextEmbedding_86165633892593
// MI455X (gfx1250) — matrix operand order repaired and checked by reading, not compiled, not run on hardware
//
#include <hip/hip_runtime.h>
#include <hip/hip_bf16.h>

// ---------------------------------------------------------------------------
// CDNA5 (gfx1250) cross-attention block, bf16 WMMA (f32 accumulate).
//   E:[8,256,1024] I:[8,4096,1024]  -> out:[8,256,2048] = concat(E, normalize(P(attn)))
// GEMMs are register-blocked 64x64 per wave: 16 WMMAs per 32-deep K step,
// 4x fragment reuse (16 b128 loads : 16 wmma).
// ---------------------------------------------------------------------------

typedef __attribute__((ext_vector_type(16))) __bf16 v16bf;
typedef __attribute__((ext_vector_type(8)))  __bf16 v8bf;
typedef __attribute__((ext_vector_type(8)))  float  v8f;

union FragBF { v16bf v; v8bf h[2]; };

#define BB_  8
#define N1_  256
#define N2_  4096
#define C_   1024
#define H_   16
#define HD_  64

static __device__ __forceinline__ v8f wmma_bf16(const FragBF& a, const FragBF& b, v8f c) {
  // D = A(16x32 bf16) * B(32x16 bf16) + C(16x16 f32)
  return __builtin_amdgcn_wmma_f32_16x16x32_bf16(false, a.v, false, b.v,
                                                 (short)0, c, false, false);
}

// ---------------------------------------------------------------- convert ---
__global__ void cvt_f32_bf16(const float* __restrict__ in, __bf16* __restrict__ out,
                             long n) {
  long i = (long)blockIdx.x * blockDim.x + threadIdx.x;
  long stride = (long)gridDim.x * blockDim.x;
  for (; i < n; i += stride) out[i] = (__bf16)in[i];
}

// ---------------------------------------------------------------- GEMM ------
// C[M x N] = A[M x Cin] * W[N x Cin]^T + bias[N]
// One wave computes a 64x64 tile: acc[4][4] 16x16 WMMA accumulators.
// mode 0: bf16 row-major out (ld = N)
// mode 1: bf16 transposed per-head: out[((b*H + h)*HD + d)*N2 + n2]
// mode 2: f32 row-major out (ld = N)
__global__ __launch_bounds__(32) void gemm_wmma_bf16(
    const __bf16* __restrict__ A, const __bf16* __restrict__ W,
    const float* __restrict__ bias, void* __restrict__ outp,
    int M, int N, int Cin, int mode)
{
  const int lane = threadIdx.x;
  const int grp  = lane >> 4;     // lane group 0/1
  const int lr   = lane & 15;
  const int m0   = blockIdx.x * 64;
  const int n0   = blockIdx.y * 64;

  const __bf16* arow[4];
  const __bf16* wrow[4];
  for (int mi = 0; mi < 4; ++mi) arow[mi] = A + (size_t)(m0 + mi * 16 + lr) * Cin;
  for (int ni = 0; ni < 4; ++ni) wrow[ni] = W + (size_t)(n0 + ni * 16 + lr) * Cin;

  v8f acc[4][4] = {};
  for (int kk = 0; kk < Cin; kk += 32) {
    FragBF af[4], bf[4];
    for (int mi = 0; mi < 4; ++mi) {
      // A layout: e0..7 -> K = kk+8*grp..+7 ; e8..15 -> K = kk+16+8*grp..+7
      af[mi].h[0] = *(const v8bf*)(arow[mi] + kk + 8 * grp);
      af[mi].h[1] = *(const v8bf*)(arow[mi] + kk + 16 + 8 * grp);
    }
    for (int ni = 0; ni < 4; ++ni) {
      // B layout: e0..7 -> K = kk+8*grp..+7 ; e8..15 -> K = kk+16+8*grp..+7
      bf[ni].h[0] = *(const v8bf*)(wrow[ni] + kk + 8 * grp);
      bf[ni].h[1] = *(const v8bf*)(wrow[ni] + kk + 16 + 8 * grp);
    }
    if (kk + 32 < Cin) {
      __builtin_prefetch(arow[0] + kk + 32, 0, 3);
      __builtin_prefetch(wrow[0] + kk + 32, 0, 3);
    }
    for (int mi = 0; mi < 4; ++mi)
      for (int ni = 0; ni < 4; ++ni)
        acc[mi][ni] = wmma_bf16(af[mi], bf[ni], acc[mi][ni]);
  }

  for (int ni = 0; ni < 4; ++ni) {
    const int   oc = n0 + ni * 16 + lr;        // output column for this lane
    const float bv = bias ? bias[oc] : 0.0f;
    if (mode == 0) {
      __bf16* out = (__bf16*)outp;
      for (int mi = 0; mi < 4; ++mi)
        for (int r = 0; r < 8; ++r) {
          int m = m0 + mi * 16 + r + 8 * grp;
          out[(size_t)m * N + oc] = (__bf16)(acc[mi][ni][r] + bv);
        }
    } else if (mode == 1) {
      __bf16* out = (__bf16*)outp;
      int hh = oc >> 6, d = oc & (HD_ - 1);
      for (int mi = 0; mi < 4; ++mi)
        for (int r = 0; r < 8; ++r) {
          int m  = m0 + mi * 16 + r + 8 * grp;
          int bb = m >> 12;                    // 4096 rows per batch
          int n2 = m & (N2_ - 1);
          out[(((size_t)bb * H_ + hh) * HD_ + d) * N2_ + n2] =
              (__bf16)(acc[mi][ni][r] + bv);
        }
    } else {
      float* out = (float*)outp;
      for (int mi = 0; mi < 4; ++mi)
        for (int r = 0; r < 8; ++r) {
          int m = m0 + mi * 16 + r + 8 * grp;
          out[(size_t)m * N + oc] = acc[mi][ni][r] + bv;
        }
    }
  }
}

// ------------------------------------------------------------- attention ----
// One wave per (b, h, 16-query-row tile). Two passes over N2:
//   pass 1: row max of S = Q K^T   (WMMA + shfl reduction)
//   pass 2: U' += exp(S - max) V, row-sum; then U = U'/rowsum.
__global__ __launch_bounds__(32) void attn_fused(
    const __bf16* __restrict__ Q, const __bf16* __restrict__ Km,
    const __bf16* __restrict__ Vt, __bf16* __restrict__ U)
{
  __shared__ __align__(16) __bf16 Pl[16 * 32];   // P tile staging (C -> A layout)

  const int lane = threadIdx.x;
  const int grp  = lane >> 4;
  const int lr   = lane & 15;
  const int mt   = blockIdx.x;     // 0..15  (query tile)
  const int hh   = blockIdx.y;     // 0..15  (head)
  const int bb   = blockIdx.z;     // 0..7   (batch)

  // Q A-fragments for the two 32-wide d slabs (HD = 64)
  const __bf16* qrow = Q + ((size_t)(bb * N1_ + mt * 16 + lr)) * C_ + hh * HD_;
  FragBF qa[2];
  for (int s = 0; s < 2; ++s) {
    qa[s].h[0] = *(const v8bf*)(qrow + s * 32 + 8 * grp);
    qa[s].h[1] = *(const v8bf*)(qrow + s * 32 + 16 + 8 * grp);
  }
  const __bf16* kbase = Km + ((size_t)bb * N2_) * C_ + hh * HD_;
  const __bf16* vbase = Vt + (((size_t)bb * H_ + hh) * HD_) * N2_;

  // ---- pass 1: row max -----------------------------------------------------
  float mx[8];
  for (int r = 0; r < 8; ++r) mx[r] = -3.0e38f;
  for (int nb = 0; nb < N2_; nb += 16) {
    v8f s = {};
    const __bf16* kp0 = kbase + (size_t)(nb + lr) * C_;
    for (int ks = 0; ks < 2; ++ks) {
      FragBF kf;
      kf.h[0] = *(const v8bf*)(kp0 + ks * 32 + 8 * grp);
      kf.h[1] = *(const v8bf*)(kp0 + ks * 32 + 16 + 8 * grp);
      s = wmma_bf16(qa[ks], kf, s);
    }
    for (int r = 0; r < 8; ++r) mx[r] = fmaxf(mx[r], s[r]);
  }
  for (int msk = 1; msk <= 8; msk <<= 1)
    for (int r = 0; r < 8; ++r)
      mx[r] = fmaxf(mx[r], __shfl_xor(mx[r], msk, 32));

  // ---- pass 2: accumulate --------------------------------------------------
  float rs[8];
  for (int r = 0; r < 8; ++r) rs[r] = 0.0f;
  v8f uacc[4] = {};   // four 16-wide d slabs of U (16x64)

  for (int nb = 0; nb < N2_; nb += 32) {
    for (int t = 0; t < 2; ++t) {
      v8f s = {};
      const __bf16* kp0 = kbase + (size_t)(nb + t * 16 + lr) * C_;
      for (int ks = 0; ks < 2; ++ks) {
        FragBF kf;
        kf.h[0] = *(const v8bf*)(kp0 + ks * 32 + 8 * grp);
        kf.h[1] = *(const v8bf*)(kp0 + ks * 32 + 16 + 8 * grp);
        s = wmma_bf16(qa[ks], kf, s);
      }
      for (int r = 0; r < 8; ++r) {
        float p = __expf(s[r] - mx[r]);
        rs[r] += p;
        Pl[(r + 8 * grp) * 32 + t * 16 + lr] = (__bf16)p;   // row-major P[16][32]
      }
    }
    __syncthreads();
    FragBF pa;  // A-fragment of P from LDS
    pa.h[0] = *(const v8bf*)(&Pl[lr * 32 + 8 * grp]);
    pa.h[1] = *(const v8bf*)(&Pl[lr * 32 + 16 + 8 * grp]);
    for (int sb = 0; sb < 4; ++sb) {
      FragBF vf;  // B-fragment from Vt: col d = sb*16+lr, K = n2 contiguous
      const __bf16* vp = vbase + (size_t)(sb * 16 + lr) * N2_ + nb + 8 * grp;
      vf.h[0] = *(const v8bf*)(vp);
      vf.h[1] = *(const v8bf*)(vp + 16);
      uacc[sb] = wmma_bf16(pa, vf, uacc[sb]);
    }
    __syncthreads();
  }

  for (int msk = 1; msk <= 8; msk <<= 1)
    for (int r = 0; r < 8; ++r)
      rs[r] += __shfl_xor(rs[r], msk, 32);

  for (int sb = 0; sb < 4; ++sb)
    for (int r = 0; r < 8; ++r) {
      int n1 = mt * 16 + r + 8 * grp;
      int d  = sb * 16 + lr;
      U[((size_t)bb * N1_ + n1) * C_ + hh * HD_ + d] = (__bf16)(uacc[sb][r] / rs[r]);
    }
}

// ------------------------------------------------------- normalize+concat ---
__global__ __launch_bounds__(256) void norm_concat(
    const float* __restrict__ T, const float* __restrict__ E, float* __restrict__ out)
{
  __shared__ float red[256];
  const int m = blockIdx.x;            // 0..2047 == (b, n1)
  const int t = threadIdx.x;
  const float* trow = T + (size_t)m * C_;
  float ss = 0.0f;
  for (int o = t; o < C_; o += 256) { float v = trow[o]; ss += v * v; }
  red[t] = ss;
  __syncthreads();
  for (int s = 128; s > 0; s >>= 1) {
    if (t < s) red[t] += red[t + s];
    __syncthreads();
  }
  const float inv = rsqrtf(red[0]);
  float* orow = out + (size_t)m * (2 * C_);
  const float* erow = E + (size_t)m * C_;
  for (int o = t; o < C_; o += 256) {
    orow[o]       = erow[o];
    orow[C_ + o]  = trow[o] * inv;
  }
}

// ---------------------------------------------------------------- launch ----
extern "C" void kernel_launch(void* const* d_in, const int* in_sizes, int n_in,
                              void* d_out, int out_size, void* d_ws, size_t ws_size,
                              hipStream_t stream)
{
  (void)in_sizes; (void)n_in; (void)out_size; (void)ws_size;

  const float* E  = (const float*)d_in[0];
  const float* I  = (const float*)d_in[1];
  const float* qw = (const float*)d_in[2];
  const float* qb = (const float*)d_in[3];
  const float* kw = (const float*)d_in[4];
  const float* kb = (const float*)d_in[5];
  const float* vw = (const float*)d_in[6];
  const float* vb = (const float*)d_in[7];
  const float* pw = (const float*)d_in[8];
  const float* pb = (const float*)d_in[9];
  float* out = (float*)d_out;

  char* ws = (char*)d_ws;
  size_t off = 0;
  auto alloc = [&](size_t bytes) -> char* {
    char* p = ws + off;
    off += (bytes + 255) & ~(size_t)255;
    return p;
  };

  const size_t nE = (size_t)BB_ * N1_ * C_;        // 2,097,152
  const size_t nI = (size_t)BB_ * N2_ * C_;        // 33,554,432
  const size_t nW = (size_t)C_ * C_;               // 1,048,576

  __bf16* Ebf = (__bf16*)alloc(nE * 2);
  __bf16* Ibf = (__bf16*)alloc(nI * 2);
  __bf16* qwb = (__bf16*)alloc(nW * 2);
  __bf16* kwb = (__bf16*)alloc(nW * 2);
  __bf16* vwb = (__bf16*)alloc(nW * 2);
  __bf16* pwb = (__bf16*)alloc(nW * 2);
  __bf16* Qbf = (__bf16*)alloc(nE * 2);
  __bf16* Kbf = (__bf16*)alloc(nI * 2);
  __bf16* Vtb = (__bf16*)alloc(nI * 2);
  __bf16* Ubf = (__bf16*)alloc(nE * 2);
  float*  Tf  = (float*) alloc(nE * 4);

  // 1) downconvert inputs + weights to bf16
  auto cvt = [&](const float* src, __bf16* dst, size_t n) {
    long blocks = (long)((n + 255) / 256); if (blocks > 4096) blocks = 4096;
    cvt_f32_bf16<<<dim3((unsigned)blocks), dim3(256), 0, stream>>>(src, dst, (long)n);
  };
  cvt(E,  Ebf, nE);
  cvt(I,  Ibf, nI);
  cvt(qw, qwb, nW);
  cvt(kw, kwb, nW);
  cvt(vw, vwb, nW);
  cvt(pw, pwb, nW);

  // 2) projections (WMMA GEMMs, 64x64 tiles per wave)
  gemm_wmma_bf16<<<dim3(BB_ * N1_ / 64, C_ / 64), dim3(32), 0, stream>>>(
      Ebf, qwb, qb, (void*)Qbf, BB_ * N1_, C_, C_, 0);
  gemm_wmma_bf16<<<dim3(BB_ * N2_ / 64, C_ / 64), dim3(32), 0, stream>>>(
      Ibf, kwb, kb, (void*)Kbf, BB_ * N2_, C_, C_, 0);
  gemm_wmma_bf16<<<dim3(BB_ * N2_ / 64, C_ / 64), dim3(32), 0, stream>>>(
      Ibf, vwb, vb, (void*)Vtb, BB_ * N2_, C_, C_, 1);   // transposed per head

  // 3) fused attention (softmax in-wave, WMMA for QK^T and PV)
  attn_fused<<<dim3(N1_ / 16, H_, BB_), dim3(32), 0, stream>>>(Qbf, Kbf, Vtb, Ubf);

  // 4) output projection (f32 out)
  gemm_wmma_bf16<<<dim3(BB_ * N1_ / 64, C_ / 64), dim3(32), 0, stream>>>(
      Ubf, pwb, pb, (void*)Tf, BB_ * N1_, C_, C_, 2);

  // 5) L2-normalize + concat with E
  norm_concat<<<dim3(BB_ * N1_), dim3(256), 0, stream>>>(Tf, E, out);
}
